// GSHLayer_44066364457374
// MI455X (gfx1250) — matrix operand order repaired and checked by reading, not compiled, not run on hardware
//
#include <hip/hip_runtime.h>

// ---------------------------------------------------------------------------
// Problem constants (match reference):  B=4, L=S=1024, D_MODEL=1024, H=16, E=64
// ---------------------------------------------------------------------------
#define CB   4
#define CL   1024
#define CS   1024
#define CH   16
#define CE   64
#define CHE  1024      /* H*E == D_MODEL */
#define CDM  1024

typedef __attribute__((ext_vector_type(16))) __bf16 v16bf;
typedef __attribute__((ext_vector_type(8)))  __bf16 v8bf;
typedef __attribute__((ext_vector_type(8)))  float  v8f;
typedef __attribute__((ext_vector_type(4)))  float  f32x4;

// ---------------------------------------------------------------------------
// CDNA5 async global->LDS copy (ASYNCcnt-tracked, no VGPR round trip).
// lds_off = byte offset within the workgroup LDS allocation (dynamic LDS
// starts at 0; these kernels have no static LDS).
// ---------------------------------------------------------------------------
static __device__ __forceinline__ void async_copy_b128(unsigned lds_off,
                                                       const void* gaddr) {
    asm volatile("global_load_async_to_lds_b128 %0, %1, off"
                 :: "v"(lds_off), "v"(gaddr) : "memory");
}
static __device__ __forceinline__ void wait_async0() {
    asm volatile("s_wait_asynccnt 0x0" ::: "memory");
}
static __device__ __forceinline__ void wait_async4() {
    asm volatile("s_wait_asynccnt 0x4" ::: "memory");
}

// ---------------------------------------------------------------------------
// WMMA helpers — CDNA5 v_wmma_f32_16x16x32_bf16, wave32 fragment layouts
// (CDNA5 ISA §7.12.2)
// ---------------------------------------------------------------------------
static __device__ __forceinline__ v8f vzero8() {
    v8f z;
#pragma unroll
    for (int i = 0; i < 8; ++i) z[i] = 0.0f;
    return z;
}

static __device__ __forceinline__ v16bf pack16(v8bf lo, v8bf hi) {
    v16bf a;
#pragma unroll
    for (int i = 0; i < 8; ++i) { a[i] = lo[i]; a[8 + i] = hi[i]; }
    return a;
}

// A-fragment (16x32, MxK) from LDS bf16 row-major [rows][ld].
static __device__ __forceinline__ v16bf frag_a_lds(const __bf16* base, int ld,
                                                   int m0, int kbase, int lane) {
    const int row = m0 + (lane & 15);
    const int g   = (lane >> 4) & 1;
    const __bf16* p = base + row * ld + kbase + g * 8;
    return pack16(*(const v8bf*)(p), *(const v8bf*)(p + 16));  // 2x ds_load_b128
}

// A-fragment with on-the-fly f32->bf16 conversion from LDS f32 row-major
// (entmax probability matrix).
static __device__ __forceinline__ v16bf frag_a_lds_f32(const float* base, int ld,
                                                       int m0, int kbase, int lane) {
    const int row = m0 + (lane & 15);
    const int g   = (lane >> 4) & 1;
    const float* p = base + row * ld + kbase;
    v16bf a;
#pragma unroll
    for (int i = 0; i < 8; ++i) a[i]     = (__bf16)p[g * 8 + i];
#pragma unroll
    for (int i = 0; i < 8; ++i) a[8 + i] = (__bf16)p[16 + g * 8 + i];
    return a;
}

// B-fragment (32x16, KxN) from K-transposed bf16 LDS tile Bt[n][k].
static __device__ __forceinline__ v16bf frag_b_ldsT(const __bf16* base, int ld,
                                                    int n0, int kbase, int lane) {
    const int n = n0 + (lane & 15);
    const int g = (lane >> 4) & 1;
    const __bf16* p = base + n * ld + kbase + g * 8;
    return pack16(*(const v8bf*)p, *(const v8bf*)(p + 16));
}

// B-fragment from global bf16 where the WMMA K-dim is contiguous in memory.
static __device__ __forceinline__ v16bf frag_b_g_bf16(const __bf16* base, int ldn,
                                                      int n0, int kbase, int lane) {
    const int n = n0 + (lane & 15);
    const int g = (lane >> 4) & 1;
    const __bf16* p = base + (long)n * ldn + kbase + g * 8;
    return pack16(*(const v8bf*)p, *(const v8bf*)(p + 16));     // 2x global b128
}

static __device__ __forceinline__ v8f wmma_bf16f32(v16bf a, v16bf b, v8f c) {
    // 8 args: (neg_a, A, neg_b, B, c_mod, C, reuse_a, reuse_b)
    return __builtin_amdgcn_wmma_f32_16x16x32_bf16(false, a, false, b,
                                                   (short)0, c, false, false);
}

// ---------------------------------------------------------------------------
// One-shot precision/layout conversion kernels (tiny vs GEMM cost)
// ---------------------------------------------------------------------------
__global__ __launch_bounds__(256) void cast_f32_to_bf16(
    const float* __restrict__ in, __bf16* __restrict__ out, long n)
{
    const long i = ((long)blockIdx.x * 256 + threadIdx.x) * 8;
    if (i >= n) return;
    f32x4 a = *(const f32x4*)(in + i);
    f32x4 b = *(const f32x4*)(in + i + 4);
    v8bf o;
#pragma unroll
    for (int j = 0; j < 4; ++j) { o[j] = (__bf16)a[j]; o[4 + j] = (__bf16)b[j]; }
    *(v8bf*)(out + i) = o;
}

// in: f32 [R][C] row-major  ->  out: bf16 [C][R]
__global__ __launch_bounds__(256) void cast_transpose_f32_to_bf16(
    const float* __restrict__ in, __bf16* __restrict__ out, int R, int C)
{
    __shared__ float tile[32][33];
    const int t = threadIdx.x, tx = t & 31, ty = t >> 5;
    const int r0 = blockIdx.y * 32, c0 = blockIdx.x * 32;
#pragma unroll
    for (int i = 0; i < 4; ++i)
        tile[ty + i * 8][tx] = in[(long)(r0 + ty + i * 8) * C + c0 + tx];
    __syncthreads();
#pragma unroll
    for (int i = 0; i < 4; ++i)
        out[(long)(c0 + ty + i * 8) * R + r0 + tx] = (__bf16)tile[tx][ty + i * 8];
}

// V [B*S][H*E] bf16  ->  VT [B][H][E][S] bf16 (per-head transpose)
__global__ __launch_bounds__(256) void transpose_v_heads(
    const __bf16* __restrict__ in, __bf16* __restrict__ out)
{
    __shared__ __bf16 tile[32][33];
    const int t = threadIdx.x, tx = t & 31, ty = t >> 5;
    const int bh = blockIdx.z, b = bh >> 4, h = bh & 15;
    const int s0 = blockIdx.x * 32, e0 = blockIdx.y * 32;
#pragma unroll
    for (int i = 0; i < 4; ++i)
        tile[ty + i * 8][tx] =
            in[(long)(b * CS + s0 + ty + i * 8) * CHE + h * CE + e0 + tx];
    __syncthreads();
#pragma unroll
    for (int i = 0; i < 4; ++i)
        out[(long)((b * CH + h) * CE + e0 + ty + i * 8) * CS + s0 + tx] =
            tile[tx][ty + i * 8];
}

// ---------------------------------------------------------------------------
// bf16 GEMM:  C[M,N] = A[M,K] @ Wt[N,K]^T + bias[N]
// Async double-buffered staging; steady-state loop body is branch-free:
//   issue 4 async b128 copies (next tile) -> s_wait_asynccnt 4 -> barrier ->
//   12x ds_load_b128 + 8x WMMA -> barrier.
// Per-thread copy pointers are computed once and stepped by BK each tile.
// ---------------------------------------------------------------------------
__global__ __launch_bounds__(256) void gemm_bf16wmma(
    const __bf16* __restrict__ A, const __bf16* __restrict__ Wt,
    const float* __restrict__ bias, float* __restrict__ Cf,
    __bf16* __restrict__ Cbf, int M, int N, int K)
{
    extern __shared__ char smem[];          // 2 x (8KB A-tile + 8KB B-tile)
    __bf16* Sb = (__bf16*)smem;

    const int t    = threadIdx.x;
    const int lane = t & 31;
    const int w    = t >> 5;
    const int m0   = blockIdx.y * 128;
    const int n0   = blockIdx.x * 128;
    const int wm   = (w >> 2) * 64;
    const int wn   = (w & 3) * 32;

    // ---- per-thread async-copy descriptors (computed once)
    const __bf16* gA[2];
    const __bf16* gW[2];
    unsigned      lA[2], lW[2];
#pragma unroll
    for (int j = 0; j < 2; ++j) {
        const int c    = t * 2 + j;          // 0..511 over 128x32 bf16 tile
        const int row  = c >> 2;
        const int koff = (c & 3) * 8;
        gA[j] = A + (long)(m0 + row) * K + koff;
        gW[j] = Wt + (long)(n0 + row) * K + koff;
        lA[j] = (unsigned)((row * 32 + koff) * 2);
        lW[j] = (unsigned)(8192 + (row * 32 + koff) * 2);
    }

    v8f acc[4][2];
#pragma unroll
    for (int mt = 0; mt < 4; ++mt)
#pragma unroll
        for (int nt = 0; nt < 2; ++nt) acc[mt][nt] = vzero8();

    unsigned bufbit = 0;                     // 0 or 0x4000 (16 KB)
    // prologue: stage tile 0 into buffer 0
#pragma unroll
    for (int j = 0; j < 2; ++j) {
        async_copy_b128(lA[j], gA[j]);
        async_copy_b128(lW[j], gW[j]);
    }

    const int nk = K / 32;

    // steady state (branch-free body) + peeled last iteration
    for (int i = 0; i < nk - 1; ++i) {
        const unsigned nb = bufbit ^ 0x4000u;
#pragma unroll
        for (int j = 0; j < 2; ++j) {
            async_copy_b128(lA[j] + nb, gA[j] + 32);
            async_copy_b128(lW[j] + nb, gW[j] + 32);
            gA[j] += 32; gW[j] += 32;
        }
        wait_async4();                       // current tile's 4 copies landed
        __syncthreads();

        const __bf16* As = Sb + (bufbit >> 1);        // bufbit/2 elements
        const __bf16* Bt = As + 4096;
        v16bf af[4];
#pragma unroll
        for (int mt = 0; mt < 4; ++mt)
            af[mt] = frag_a_lds(As, 32, wm + mt * 16, 0, lane);
        v16bf bf2[2];
#pragma unroll
        for (int nt = 0; nt < 2; ++nt)
            bf2[nt] = frag_b_ldsT(Bt, 32, wn + nt * 16, 0, lane);
#pragma unroll
        for (int mt = 0; mt < 4; ++mt)
#pragma unroll
            for (int nt = 0; nt < 2; ++nt)
                acc[mt][nt] = wmma_bf16f32(af[mt], bf2[nt], acc[mt][nt]);
        __syncthreads();
        bufbit = nb;
    }

    // last tile
    wait_async0();
    __syncthreads();
    {
        const __bf16* As = Sb + (bufbit >> 1);
        const __bf16* Bt = As + 4096;
        v16bf af[4];
#pragma unroll
        for (int mt = 0; mt < 4; ++mt)
            af[mt] = frag_a_lds(As, 32, wm + mt * 16, 0, lane);
        v16bf bf2[2];
#pragma unroll
        for (int nt = 0; nt < 2; ++nt)
            bf2[nt] = frag_b_ldsT(Bt, 32, wn + nt * 16, 0, lane);
#pragma unroll
        for (int mt = 0; mt < 4; ++mt)
#pragma unroll
            for (int nt = 0; nt < 2; ++nt)
                acc[mt][nt] = wmma_bf16f32(af[mt], bf2[nt], acc[mt][nt]);
    }

    // epilogue: D layout  M = r + 8*(lane/16), N = lane%16
    const int g  = lane >> 4;
    const int nn = lane & 15;
#pragma unroll
    for (int mt = 0; mt < 4; ++mt)
#pragma unroll
        for (int nt = 0; nt < 2; ++nt)
#pragma unroll
            for (int r = 0; r < 8; ++r) {
                const int row = m0 + wm + mt * 16 + r + 8 * g;
                const int col = n0 + wn + nt * 16 + nn;
                const float v = acc[mt][nt][r] + bias[col];
                if (Cf)  Cf[(long)row * N + col] = v;
                if (Cbf) Cbf[(long)row * N + col] = (__bf16)v;
            }
}

// ---------------------------------------------------------------------------
// Fused attention: per workgroup = one (b, h, 32-row L tile).
//   stage 1: Q tile -> LDS via async copy (bf16)
//   stage 2: scores (32x1024) via WMMA -> LDS f32 (x 1/(2*sqrt(E)))
//   stage 3: entmax-1.5 per row via monotone bisection on tau
//   stage 4: P(32x1024) @ V_h via WMMA (V pre-transposed [E][S])
// ---------------------------------------------------------------------------
#define QT_OFF 131072   /* byte offset of Q tile in dynamic LDS */

__global__ __launch_bounds__(256) void attn_entmax_fused(
    const __bf16* __restrict__ Q, const __bf16* __restrict__ Kh,
    const __bf16* __restrict__ VT, __bf16* __restrict__ O)
{
    extern __shared__ char smem[];
    float*  sc     = (float*)smem;                 // [32][1024] f32   128 KB
    __bf16* qt     = (__bf16*)(sc + 32 * 1024);    // [32][64]  bf16     4 KB
    float*  red    = (float*)(qt + 32 * 64);       // [32][8]
    float*  rstat  = red + 32 * 8;                 // [32] max / inv-sum
    float*  rstat2 = rstat + 32;                   // [32] f(tau)

    const int t    = threadIdx.x;
    const int lane = t & 31;
    const int w    = t >> 5;
    const int l0   = blockIdx.x * 32;
    const int h    = blockIdx.y;
    const int b    = blockIdx.z;

    const __bf16* Qb  = Q + ((long)b * CL) * CHE + h * CE;
    const __bf16* Kb  = Kh + ((long)b * CS) * CHE + h * CE;
    const __bf16* VTb = VT + ((long)(b * CH + h) * CE) * CS;   // [E][S]
    __bf16*       Ob  = O + ((long)b * CL) * CHE + h * CE;

    // ---- stage 1: Q tile (32x64 bf16 = 4KB) -> LDS, one async b128 / thread
    {
        const int row = t >> 3;
        const int e0  = (t & 7) * 8;
        async_copy_b128((unsigned)(QT_OFF + (row * 64 + e0) * 2),
                        Qb + (long)(l0 + row) * CHE + e0);
        wait_async0();
    }
    __syncthreads();

    // ---- stage 2: scores; wave w owns s-range [w*128, w*128+128)
    const float kscale = 0.0625f;                  // (1/sqrt(64)) * 0.5
    v16bf aq[2][2];
#pragma unroll
    for (int mt = 0; mt < 2; ++mt)
#pragma unroll
        for (int ks = 0; ks < 2; ++ks)
            aq[mt][ks] = frag_a_lds(qt, 64, mt * 16, ks * 32, lane);

    for (int j = 0; j < 8; ++j) {
        const int s0 = w * 128 + j * 16;
        v8f a0 = vzero8(), a1 = vzero8();
#pragma unroll
        for (int ks = 0; ks < 2; ++ks) {
            v16bf bk = frag_b_g_bf16(Kb, CHE, s0, ks * 32, lane);
            a0 = wmma_bf16f32(aq[0][ks], bk, a0);
            a1 = wmma_bf16f32(aq[1][ks], bk, a1);
        }
        const int g  = lane >> 4;
        const int nn = lane & 15;
#pragma unroll
        for (int rr = 0; rr < 8; ++rr) {
            sc[(rr + 8 * g) * 1024 + s0 + nn]      = a0[rr] * kscale;
            sc[(16 + rr + 8 * g) * 1024 + s0 + nn] = a1[rr] * kscale;
        }
    }
    __syncthreads();

    // ---- stage 3: entmax-1.5 per row (8 threads/row, 128 elems each)
    const int r  = t >> 3;
    const int q8 = t & 7;
    float* myrow = sc + r * 1024 + q8 * 128;

    float mx = -1e30f;
#pragma unroll 8
    for (int i = 0; i < 128; ++i) mx = fmaxf(mx, myrow[i]);
    red[r * 8 + q8] = mx;
    __syncthreads();
    if (q8 == 0) {
        float m = red[r * 8];
#pragma unroll
        for (int k = 1; k < 8; ++k) m = fmaxf(m, red[r * 8 + k]);
        rstat[r] = m;
    }
    __syncthreads();
    const float m = rstat[r];

    float lo = m - 1.0f, hi = m - 0.03125f;
    for (int it = 0; it < 25; ++it) {
        const float tau = 0.5f * (lo + hi);
        float p = 0.0f;
#pragma unroll 8
        for (int i = 0; i < 128; ++i) {
            float d = fmaxf(myrow[i] - tau, 0.0f);
            p = fmaf(d, d, p);
        }
        red[r * 8 + q8] = p;
        __syncthreads();
        if (q8 == 0) {
            float f = red[r * 8];
#pragma unroll
            for (int k = 1; k < 8; ++k) f += red[r * 8 + k];
            rstat2[r] = f;
        }
        __syncthreads();
        const float f = rstat2[r];
        if (f >= 1.0f) lo = tau; else hi = tau;   // uniform across row threads
    }
    const float tau = 0.5f * (lo + hi);

    float psum = 0.0f;
#pragma unroll 8
    for (int i = 0; i < 128; ++i) {
        float d = fmaxf(myrow[i] - tau, 0.0f);
        float pv = d * d;
        myrow[i] = pv;
        psum += pv;
    }
    red[r * 8 + q8] = psum;
    __syncthreads();
    if (q8 == 0) {
        float s = red[r * 8];
#pragma unroll
        for (int k = 1; k < 8; ++k) s += red[r * 8 + k];
        rstat[r] = 1.0f / fmaxf(s, 1e-30f);
    }
    __syncthreads();

    // ---- stage 4: P @ V_h; wave w -> (mt = w&1, e-tile n0 = (w>>1)*16)
    {
        const int mt = w & 1;
        const int n0 = (w >> 1) * 16;
        v8f acc = vzero8();
        for (int ks = 0; ks < 32; ++ks) {
            v16bf pa = frag_a_lds_f32(sc, 1024, mt * 16, ks * 32, lane);
            v16bf bv = frag_b_g_bf16(VTb, CS, n0, ks * 32, lane);  // contiguous
            acc = wmma_bf16f32(pa, bv, acc);
        }
        const int g  = lane >> 4;
        const int nn = lane & 15;
#pragma unroll
        for (int rr = 0; rr < 8; ++rr) {
            const int row = mt * 16 + rr + 8 * g;
            Ob[(long)(l0 + row) * CHE + n0 + nn] = (__bf16)(acc[rr] * rstat[row]);
        }
    }
}

// ---------------------------------------------------------------------------
// Host launcher
// ---------------------------------------------------------------------------
extern "C" void kernel_launch(void* const* d_in, const int* in_sizes, int n_in,
                              void* d_out, int out_size, void* d_ws, size_t ws_size,
                              hipStream_t stream) {
    const float* queries = (const float*)d_in[0];
    const float* keys    = (const float*)d_in[1];
    const float* values  = (const float*)d_in[2];
    const float* Wq      = (const float*)d_in[3];
    const float* bq      = (const float*)d_in[4];
    const float* Wk      = (const float*)d_in[5];
    const float* bk      = (const float*)d_in[6];
    const float* Wv      = (const float*)d_in[7];
    const float* bv      = (const float*)d_in[8];
    const float* Wo      = (const float*)d_in[9];
    const float* bo      = (const float*)d_in[10];
    float* out = (float*)d_out;

    const int  M  = CB * CL;                       // 4096
    const int  DM = CDM;                           // 1024
    const long actN = (long)M * DM;                // 4M elements
    const long wN   = (long)DM * DM;               // 1M elements

    char* ws = (char*)d_ws;
    size_t off = 0;
    __bf16* qIn = (__bf16*)(ws + off); off += actN * 2;    // 8 MB
    __bf16* kIn = (__bf16*)(ws + off); off += actN * 2;
    __bf16* vIn = (__bf16*)(ws + off); off += actN * 2;
    __bf16* WqT = (__bf16*)(ws + off); off += wN * 2;      // 2 MB each
    __bf16* WkT = (__bf16*)(ws + off); off += wN * 2;
    __bf16* WvT = (__bf16*)(ws + off); off += wN * 2;
    __bf16* WoT = (__bf16*)(ws + off); off += wN * 2;
    __bf16* Qp  = (__bf16*)(ws + off); off += actN * 2;
    __bf16* Kp  = (__bf16*)(ws + off); off += actN * 2;
    __bf16* Vt  = (__bf16*)(ws + off); off += actN * 2;
    __bf16* Vp  = (__bf16*)(ws + off); off += actN * 2;
    __bf16* VpT = (__bf16*)(ws + off); off += actN * 2;
    __bf16* AO  = (__bf16*)(ws + off); off += actN * 2;    // total ~80 MB

    dim3 blk(256);

    // ---- precision / layout conversion (one-shot, tiny)
    cast_f32_to_bf16<<<dim3(actN / (256 * 8)), blk, 0, stream>>>(queries, qIn, actN);
    cast_f32_to_bf16<<<dim3(actN / (256 * 8)), blk, 0, stream>>>(keys,    kIn, actN);
    cast_f32_to_bf16<<<dim3(actN / (256 * 8)), blk, 0, stream>>>(values,  vIn, actN);
    dim3 gt(DM / 32, DM / 32);
    cast_transpose_f32_to_bf16<<<gt, blk, 0, stream>>>(Wq, WqT, DM, DM);
    cast_transpose_f32_to_bf16<<<gt, blk, 0, stream>>>(Wk, WkT, DM, DM);
    cast_transpose_f32_to_bf16<<<gt, blk, 0, stream>>>(Wv, WvT, DM, DM);
    cast_transpose_f32_to_bf16<<<gt, blk, 0, stream>>>(Wo, WoT, DM, DM);

    // ---- projection GEMMs (bf16 WMMA, async double-buffered staging)
    dim3 gg(DM / 128, M / 128);
    const size_t ldsG = 2 * 16384;                 // 32 KB (double buffer)
    gemm_bf16wmma<<<gg, blk, ldsG, stream>>>(qIn, WqT, bq, nullptr, Qp, M, DM, DM);
    gemm_bf16wmma<<<gg, blk, ldsG, stream>>>(kIn, WkT, bk, nullptr, Kp, M, DM, DM);
    gemm_bf16wmma<<<gg, blk, ldsG, stream>>>(vIn, WkT, bk, nullptr, Vt, M, DM, DM);
    gemm_bf16wmma<<<gg, blk, ldsG, stream>>>(Vt,  WvT, bv, nullptr, Vp, M, DM, DM);

    // per-head transpose of V: [B*S][H*E] -> [B][H][E][S]
    transpose_v_heads<<<dim3(CS / 32, CE / 32, CB * CH), blk, 0, stream>>>(Vp, VpT);

    // ---- fused scores -> entmax15 -> P@V
    dim3 ga(CL / 32, CH, CB);
    const size_t ldsA = (size_t)32 * 1024 * 4     // score stripe (f32)
                      + (size_t)32 * 64 * 2       // Q tile (bf16)
                      + (size_t)32 * 8 * 4        // reduction scratch
                      + (size_t)32 * 4 * 2;       // row stats
    attn_entmax_fused<<<ga, blk, ldsA, stream>>>(Qp, Kp, VpT, AO);

    // ---- output projection (f32 result)
    gemm_bf16wmma<<<gg, blk, ldsG, stream>>>(AO, WoT, bo, out, nullptr, M, DM, DM);
}
